// ODConvBlock_87849261072426
// MI455X (gfx1250) — matrix operand order repaired and checked by reading, not compiled, not run on hardware
//
#include <hip/hip_runtime.h>
#include <hip/hip_bf16.h>

typedef __attribute__((ext_vector_type(16))) _Float16 v16h;
typedef __attribute__((ext_vector_type(8)))  float    v8f;

struct alignas(32) HalfFrag { uint4 u[2]; };

// ---------------------------------------------------------------------------
// Stage 1: global average pool.  pooled[b*64+ci] = mean(x[b,ci,:,:])
// One block per (b,ci) plane (64KB contiguous), float4 loads + tree reduce.
// ---------------------------------------------------------------------------
__global__ void __launch_bounds__(256) pool_kernel(const float* __restrict__ x,
                                                   float* __restrict__ pooled) {
    const int plane = blockIdx.x;                       // b*64 + ci, 2048 planes
    const float4* p = (const float4*)(x + (size_t)plane * 16384);
    float s = 0.f;
    for (int i = threadIdx.x; i < 4096; i += 256) {
        float4 v = p[i];
        s += v.x + v.y + v.z + v.w;
    }
    __shared__ float red[256];
    red[threadIdx.x] = s;
    __syncthreads();
    for (int off = 128; off > 0; off >>= 1) {
        if (threadIdx.x < off) red[threadIdx.x] += red[threadIdx.x + off];
        __syncthreads();
    }
    if (threadIdx.x == 0) pooled[plane] = red[0] * (1.f / 16384.f);
}

// ---------------------------------------------------------------------------
// Stage 2: attention trunk + 4 heads. One wave, one thread per batch sample.
// ---------------------------------------------------------------------------
__global__ void attn_kernel(const float* __restrict__ pooled,
                            const float* __restrict__ prep_w,
                            const float* __restrict__ g,  const float* __restrict__ be,
                            const float* __restrict__ mu, const float* __restrict__ var,
                            const float* __restrict__ spw, const float* __restrict__ spb,
                            const float* __restrict__ chw, const float* __restrict__ chb,
                            const float* __restrict__ fw,  const float* __restrict__ fb,
                            const float* __restrict__ kw,  const float* __restrict__ kb,
                            float* __restrict__ sp, float* __restrict__ cha,
                            float* __restrict__ fa, float* __restrict__ ka) {
    const int b = threadIdx.x;
    if (b >= 32) return;
    float att[16];
    for (int a = 0; a < 16; ++a) {
        float s = 0.f;
        for (int i = 0; i < 64; ++i) s += pooled[b * 64 + i] * prep_w[a * 64 + i];
        s = (s - mu[a]) * rsqrtf(var[a] + 1e-5f) * g[a] + be[a];
        att[a] = fmaxf(s, 0.f);
    }
    for (int j = 0; j < 9; ++j) {
        float s = spb[j];
        for (int a = 0; a < 16; ++a) s += att[a] * spw[j * 16 + a];
        sp[b * 9 + j] = 1.f / (1.f + __expf(-s));
    }
    for (int i = 0; i < 64; ++i) {
        float s = chb[i];
        for (int a = 0; a < 16; ++a) s += att[a] * chw[i * 16 + a];
        cha[b * 64 + i] = 1.f / (1.f + __expf(-s));
    }
    for (int o = 0; o < 64; ++o) {
        float s = fb[o];
        for (int a = 0; a < 16; ++a) s += att[a] * fw[o * 16 + a];
        fa[b * 64 + o] = 1.f / (1.f + __expf(-s));
    }
    float lg[4], mx = -1e30f, den = 0.f;
    for (int k = 0; k < 4; ++k) {
        float s = kb[k];
        for (int a = 0; a < 16; ++a) s += att[a] * kw[k * 16 + a];
        lg[k] = s; mx = fmaxf(mx, s);
    }
    for (int k = 0; k < 4; ++k) { lg[k] = __expf(lg[k] - mx); den += lg[k]; }
    for (int k = 0; k < 4; ++k) ka[b * 4 + k] = lg[k] / den;
}

// ---------------------------------------------------------------------------
// Stage 3: weight synthesis, all attention factors folded in, written DIRECTLY
// in per-lane WMMA A-fragment order (contiguous k order per lane half, same as the B reads):
//   Asw[((b*4+msub)*18 + s)*32*16 + lane*16 + e]  (f16)
//   K(e,half) = 16*half + e
//   global k = s*32 + K ; ordered as (tap, i):  i = (s&1)*32 + K, tap = s/2
// ---------------------------------------------------------------------------
__global__ void __launch_bounds__(256) wsyn_kernel(const float* __restrict__ kernels,
                                                   const float* __restrict__ sp,
                                                   const float* __restrict__ cha,
                                                   const float* __restrict__ fa,
                                                   const float* __restrict__ ka,
                                                   _Float16* __restrict__ Asw) {
    const int gid  = blockIdx.x * 256 + threadIdx.x;    // 1,179,648 total
    const int e    = gid & 15;
    const int lane = (gid >> 4) & 31;
    const int s    = (gid >> 9) % 18;
    const int t    = gid / 9216;
    const int msub = t & 3;
    const int b    = t >> 2;

    const int half = lane >> 4;
    const int j    = e >> 1;
    const int K    = 16 * half + e;
    const int i    = ((s & 1) << 5) + K;                // input channel 0..63
    const int o    = (msub << 4) + (lane & 15);         // output channel 0..63
    const int tap  = s >> 1;                            // 3x3 tap 0..8

    float acc = 0.f;
#pragma unroll
    for (int k = 0; k < 4; ++k)
        acc += ka[b * 4 + k] * kernels[((k * 64 + o) * 64 + i) * 9 + tap];
    const float v = fa[b * 64 + o] * cha[b * 64 + i] * sp[b * 9 + tap] * acc;
    Asw[gid] = (_Float16)v;
}

// ---------------------------------------------------------------------------
// Stage 4: per-sample implicit-GEMM 3x3 conv on the WMMA pipe.
// Block = (row y, sample b), 256 threads = 8 waves in a 4(M) x 2(N) grid.
// LDS caches 3 padded rows, channel-fastest (stride 72 halves -> 16B-aligned
// ds_load_b128 B-fragment loads). A-fragments are two global b128 per lane.
// ---------------------------------------------------------------------------
#define CHS 72          // channel stride in LDS (64 data + pad, mult of 8)
#define XC_HALFS (3 * 130 * CHS)

__global__ void __launch_bounds__(256) conv_kernel(const float* __restrict__ x,
                                                   const _Float16* __restrict__ Asw,
                                                   float* __restrict__ out) {
    __shared__ _Float16 xc[XC_HALFS];
    const int y   = blockIdx.x;                         // image row 0..127
    const int b   = blockIdx.y;                         // sample 0..31
    const int tid = threadIdx.x;

    // zero (covers border columns 0/129 and out-of-image rows)
    for (int idx = tid; idx < XC_HALFS; idx += 256) xc[idx] = (_Float16)0.f;
    __syncthreads();

    // fill: rows y-1..y+1, all 64 channels, f32 -> f16, coalesced in ximg
    const float* xb = x + (size_t)b * 64 * 16384;
    for (int idx = tid; idx < 3 * 64 * 128; idx += 256) {
        const int xi  = idx & 127;
        const int ch  = (idx >> 7) & 63;
        const int r   = idx >> 13;
        const int row = y + r - 1;
        if (row >= 0 && row < 128) {
            const float v = xb[ch * 16384 + row * 128 + xi];
            xc[(r * 130 + xi + 1) * CHS + ch] = (_Float16)v;
        }
    }
    __syncthreads();

    const int w    = tid >> 5;
    const int ln   = tid & 31;
    const int msub = w & 3;                             // M sub-tile (out-ch /16)
    const int ngrp = w >> 2;                            // N half (64 px each)
    const int half = ln >> 4;
    const int nx   = ln & 15;

    v8f acc[4] = {v8f{}, v8f{}, v8f{}, v8f{}};

    const _Float16* ab = Asw + ((size_t)(b * 4 + msub) * 576 + ln) * 16;

    for (int s = 0; s < 18; ++s) {
        const int tap   = s >> 1;
        const int ky    = tap / 3;
        const int kx    = tap - ky * 3;
        const int ibase = (s & 1) << 5;

        // A fragment: lane's 16 halves are contiguous (pre-swizzled), 32B
        HalfFrag at;
        const uint4* ap = (const uint4*)(ab + (size_t)s * 512);
        at.u[0] = ap[0];
        at.u[1] = ap[1];
        const v16h afrag = *(const v16h*)&at;

#pragma unroll
        for (int nt = 0; nt < 4; ++nt) {
            const int n   = ngrp * 64 + nt * 16 + nx;   // pixel x coordinate
            const int off = (ky * 130 + n + kx) * CHS + ibase + 16 * half;
            HalfFrag bt;
            const uint4* bp = (const uint4*)&xc[off];   // 16B aligned
            bt.u[0] = bp[0];
            bt.u[1] = bp[1];
            const v16h bfrag = *(const v16h*)&bt;
            acc[nt] = __builtin_amdgcn_wmma_f32_16x16x32_f16(
                false, afrag, false, bfrag, (short)0, acc[nt], false, false);
        }
    }

    // epilogue: fa/cha already folded into weights, plain f32 store
    float* ob = out + (size_t)b * 64 * 16384 + y * 128;
#pragma unroll
    for (int nt = 0; nt < 4; ++nt) {
        const int n = ngrp * 64 + nt * 16 + nx;
#pragma unroll
        for (int v = 0; v < 8; ++v) {
            const int o = msub * 16 + v + 8 * half;     // C/D layout: M = v + 8*half
            ob[(size_t)o * 16384 + n] = acc[nt][v];
        }
    }
}

// ---------------------------------------------------------------------------
extern "C" void kernel_launch(void* const* d_in, const int* in_sizes, int n_in,
                              void* d_out, int out_size, void* d_ws, size_t ws_size,
                              hipStream_t stream) {
    const float* x      = (const float*)d_in[0];
    const float* prep_w = (const float*)d_in[1];
    const float* bn_g   = (const float*)d_in[2];
    const float* bn_b   = (const float*)d_in[3];
    const float* bn_m   = (const float*)d_in[4];
    const float* bn_v   = (const float*)d_in[5];
    const float* spw    = (const float*)d_in[6];
    const float* spb    = (const float*)d_in[7];
    const float* chw    = (const float*)d_in[8];
    const float* chb    = (const float*)d_in[9];
    const float* fw     = (const float*)d_in[10];
    const float* fb     = (const float*)d_in[11];
    const float* kw     = (const float*)d_in[12];
    const float* kb     = (const float*)d_in[13];
    const float* kern   = (const float*)d_in[14];

    float* ws     = (float*)d_ws;
    float* pooled = ws;                 // 2048 f32
    float* sp     = ws + 2048;          // 288
    float* cha    = ws + 2048 + 288;    // 2048
    float* fa     = cha + 2048;         // 2048
    float* ka     = fa + 2048;          // 128
    _Float16* Asw = (_Float16*)((char*)d_ws + 32768);   // 1,179,648 f16 = 2.25MB

    pool_kernel<<<2048, 256, 0, stream>>>(x, pooled);
    attn_kernel<<<1, 32, 0, stream>>>(pooled, prep_w, bn_g, bn_b, bn_m, bn_v,
                                      spw, spb, chw, chb, fw, fb, kw, kb,
                                      sp, cha, fa, ka);
    wsyn_kernel<<<4608, 256, 0, stream>>>(kern, sp, cha, fa, ka, Asw);
    conv_kernel<<<dim3(128, 32), 256, 0, stream>>>(x, Asw, (float*)d_out);
}
